// GroundingDinoDecoderLayer_64123861729333
// MI455X (gfx1250) — matrix operand order repaired and checked by reading, not compiled, not run on hardware
//
#include <hip/hip_runtime.h>
#include <hip/hip_bf16.h>
#include <cstdint>
#include <cstddef>

typedef _Float16 f16;
typedef _Float16 v16h __attribute__((ext_vector_type(16)));
typedef _Float16 v8h  __attribute__((ext_vector_type(8)));
typedef float    v8f  __attribute__((ext_vector_type(8)));

#define DMODEL 256
#define LN_EPS 1e-5f

// ---------------------------------------------------------------------------
// gfx1250 async global->LDS copy (16B per lane) + ASYNCcnt drain.
// LDS operand is the 32-bit LDS byte offset (low 32 bits of the generic ptr).
// ---------------------------------------------------------------------------
__device__ __forceinline__ void async_ld_b128(void* lds_ptr, const void* gptr) {
    unsigned loff = (unsigned)(uintptr_t)lds_ptr;
    unsigned long long ga = (unsigned long long)(uintptr_t)gptr;
    asm volatile("global_load_async_to_lds_b128 %0, %1, off"
                 :: "v"(loff), "v"(ga) : "memory");
}
__device__ __forceinline__ void wait_async0() {
    asm volatile("s_wait_asynccnt 0x0" ::: "memory");
}

// ---------------------------------------------------------------------------
// Row-of-16 butterfly reductions (lanes 0-15 / 16-31 independent).
// Prefer v_permlane16_b32 (pure VALU) over ds_bpermute-based shuffles.
// ---------------------------------------------------------------------------
#if __has_builtin(__builtin_amdgcn_permlane16)
__device__ __forceinline__ float perm16_xor(float v, unsigned lo, unsigned hi) {
    return __uint_as_float(__builtin_amdgcn_permlane16(
        __float_as_uint(v), __float_as_uint(v), lo, hi, false, false));
}
__device__ __forceinline__ float redmax16(float v) {
    v = fmaxf(v, perm16_xor(v, 0x67452301u, 0xEFCDAB89u));   // xor 1
    v = fmaxf(v, perm16_xor(v, 0x54761032u, 0xDCFE98BAu));   // xor 2
    v = fmaxf(v, perm16_xor(v, 0x32107654u, 0xBA98FEDCu));   // xor 4
    v = fmaxf(v, perm16_xor(v, 0xFEDCBA98u, 0x76543210u));   // xor 8
    return v;
}
__device__ __forceinline__ float redsum16(float v) {
    v += perm16_xor(v, 0x67452301u, 0xEFCDAB89u);
    v += perm16_xor(v, 0x54761032u, 0xDCFE98BAu);
    v += perm16_xor(v, 0x32107654u, 0xBA98FEDCu);
    v += perm16_xor(v, 0xFEDCBA98u, 0x76543210u);
    return v;
}
#else
__device__ __forceinline__ float redmax16(float v) {
    #pragma unroll
    for (int d = 1; d < 16; d <<= 1) v = fmaxf(v, __shfl_xor(v, d, 32));
    return v;
}
__device__ __forceinline__ float redsum16(float v) {
    #pragma unroll
    for (int d = 1; d < 16; d <<= 1) v += __shfl_xor(v, d, 32);
    return v;
}
#endif

// ---------------------------------------------------------------------------
// f32 -> f16 elementwise conversion
// ---------------------------------------------------------------------------
__global__ void f32_to_f16_kernel(const float* __restrict__ src,
                                  f16* __restrict__ dst, int n) {
    int i = blockIdx.x * blockDim.x + threadIdx.x;
    if (i < n) dst[i] = (f16)src[i];
}

// ---------------------------------------------------------------------------
// Tiled WMMA GEMM:  out[M,N] = A[M,K](f16) @ W[K,N](f32 -> f16) + bias
// Block: 128 threads = 4 waves; block tile 64x64; each wave owns a 32x32
// quadrant (2x2 WMMA accumulators). K staged through LDS in chunks of 32.
// A tile staged with async global->LDS B128 copies (ASYNCcnt path).
// Ws tile stored column-major so B-fragment reads are 16B vector LDS loads.
// Requires M%64==0, N%64==0, K%32==0 (true for all calls here).
// ---------------------------------------------------------------------------
template<bool OUTF16, bool RELU>
__global__ __launch_bounds__(128)
void gemm_kernel(const f16* __restrict__ A, const float* __restrict__ W,
                 const float* __restrict__ bias, void* __restrict__ out,
                 int M, int N, int K)
{
    __shared__ __align__(16) f16 As[64 * 40];   // [row][k]  row stride 40
    __shared__ __align__(16) f16 Ws[64 * 40];   // [col][k]  col stride 40

    const int tid  = threadIdx.x;
    const int lane = tid & 31;
    const int wid  = tid >> 5;
    const int wm   = wid >> 1;          // wave row in block tile
    const int wn   = wid & 1;           // wave col in block tile
    const int r    = lane & 15;
    const int hi   = lane >> 4;
    const int m0   = blockIdx.y * 64;
    const int n0   = blockIdx.x * 64;

    v8f acc[2][2];
    #pragma unroll
    for (int tm = 0; tm < 2; ++tm)
        #pragma unroll
        for (int tn = 0; tn < 2; ++tn)
            acc[tm][tn] = (v8f){};

    const int arow  = tid >> 1;         // 0..63
    const int apart = tid & 1;          // which 16-half chunk of the 32-k row

    for (int k0 = 0; k0 < K; k0 += 32) {
        // ---- stage A tile (64x32 f16) via async global->LDS --------------
        {
            const f16* gsrc = A + (size_t)(m0 + arow) * K + k0 + apart * 16;
            f16* ldst = &As[arow * 40 + apart * 16];
            async_ld_b128(ldst,     gsrc);
            async_ld_b128(ldst + 8, gsrc + 8);
        }
        // ---- stage W tile (32k x 64n), f32 -> f16, column-major ----------
        #pragma unroll
        for (int i = 0; i < 4; ++i) {
            int f   = i * 128 + tid;      // 0..511 float4s
            int row = f >> 4;             // k     : 0..31
            int c4  = f & 15;             // n/4   : 0..15
            const float4 w4 = *reinterpret_cast<const float4*>(
                W + (size_t)(k0 + row) * N + n0 + c4 * 4);
            Ws[(c4 * 4 + 0) * 40 + row] = (f16)w4.x;
            Ws[(c4 * 4 + 1) * 40 + row] = (f16)w4.y;
            Ws[(c4 * 4 + 2) * 40 + row] = (f16)w4.z;
            Ws[(c4 * 4 + 3) * 40 + row] = (f16)w4.w;
        }
        wait_async0();
        __syncthreads();

        // ---- load fragments (ISA 7.12.2 layouts), 16B LDS vector loads ---
        v16h a[2], b[2];
        #pragma unroll
        for (int tm = 0; tm < 2; ++tm) {
            const f16* ap = &As[(wm * 32 + tm * 16 + r) * 40 + hi * 8];
            v8h lo  = *reinterpret_cast<const v8h*>(ap);        // K {0..7}/{8..15}
            v8h hi8 = *reinterpret_cast<const v8h*>(ap + 16);   // K {16..23}/{24..31}
            a[tm] = __builtin_shufflevector(lo, hi8, 0, 1, 2, 3, 4, 5, 6, 7,
                                            8, 9, 10, 11, 12, 13, 14, 15);
        }
        #pragma unroll
        for (int tn = 0; tn < 2; ++tn) {
            const f16* wp = &Ws[(wn * 32 + tn * 16 + r) * 40 + hi * 8];
            v8h b0 = *reinterpret_cast<const v8h*>(wp);         // k = hi*8 + 0..7
            v8h b1 = *reinterpret_cast<const v8h*>(wp + 16);    // k = 16 + hi*8 + 0..7
            b[tn] = __builtin_shufflevector(b0, b1, 0, 1, 2, 3, 4, 5, 6, 7,
                                            8, 9, 10, 11, 12, 13, 14, 15);
        }
        #pragma unroll
        for (int tm = 0; tm < 2; ++tm)
            #pragma unroll
            for (int tn = 0; tn < 2; ++tn)
                acc[tm][tn] = __builtin_amdgcn_wmma_f32_16x16x32_f16(
                    false, a[tm], false, b[tn], (short)0, acc[tm][tn],
                    false, false);
        __syncthreads();
    }

    // ---- epilogue: bias (+ReLU), store f16 or f32 -------------------------
    #pragma unroll
    for (int tm = 0; tm < 2; ++tm) {
        #pragma unroll
        for (int tn = 0; tn < 2; ++tn) {
            const int col = n0 + wn * 32 + tn * 16 + r;
            const float bv = bias[col];
            #pragma unroll
            for (int i = 0; i < 8; ++i) {
                const int row = m0 + wm * 32 + tm * 16 + hi * 8 + i;
                float v = acc[tm][tn][i] + bv;
                if (RELU) v = fmaxf(v, 0.0f);
                if (OUTF16)
                    ((f16*)out)[(size_t)row * N + col] = (f16)v;
                else
                    ((float*)out)[(size_t)row * N + col] = v;
            }
        }
    }
}

// ---------------------------------------------------------------------------
// Flash attention, one wave per (b, h, 32 query rows) -> 2 q-tiles share
// every K/V fragment (2x arithmetic intensity on the K/V stream).
// Q/K/V are f16 in (b, s, h*32+d) layout (GEMM output, row stride 256).
// dh = 32 == WMMA K, so scores need exactly one WMMA per 16x16 tile.
// mask is additive (B,1,1,Sk). Online softmax over Sk in chunks of 32.
// ---------------------------------------------------------------------------
__global__ __launch_bounds__(32)
void attn_kernel(const f16* __restrict__ Q, const f16* __restrict__ Km,
                 const f16* __restrict__ Vm, const float* __restrict__ mask,
                 f16* __restrict__ ctx, int Sq, int Sk)
{
    const int lane = threadIdx.x;
    const int r    = lane & 15;
    const int hi   = lane >> 4;
    const int q0   = blockIdx.x * 32;
    const int h    = blockIdx.y;
    const int b    = blockIdx.z;
    const float scale = 0.1767766952966369f;   // 1/sqrt(32)

    __shared__ __align__(16) f16 P[2 * 16 * 32];   // probs staging per q-tile

    // Q fragments held in registers for the whole Sk loop
    v16h aq[2];
    #pragma unroll
    for (int qt = 0; qt < 2; ++qt) {
        const f16* qp = Q + (size_t)(b * Sq + q0 + qt * 16 + r) * DMODEL + h * 32;
        v8h lo  = *reinterpret_cast<const v8h*>(qp + hi * 8);
        v8h hi8 = *reinterpret_cast<const v8h*>(qp + 16 + hi * 8);
        aq[qt] = __builtin_shufflevector(lo, hi8, 0, 1, 2, 3, 4, 5, 6, 7,
                                         8, 9, 10, 11, 12, 13, 14, 15);
    }

    float m_i[2][8], l_i[2][8];
    v8f o[2][2];
    #pragma unroll
    for (int qt = 0; qt < 2; ++qt) {
        #pragma unroll
        for (int i = 0; i < 8; ++i) { m_i[qt][i] = -1e30f; l_i[qt][i] = 0.0f; }
        o[qt][0] = (v8f){};
        o[qt][1] = (v8f){};
    }

    const float* mrow = mask + (size_t)b * Sk;

    for (int k0 = 0; k0 < Sk; k0 += 32) {
        // ---- K fragments (shared by both q-tiles) -------------------------
        v16h kb0, kb1;
        {
            const f16* kp = Km + (size_t)(b * Sk + k0 + r) * DMODEL
                            + h * 32 + hi * 8;
            v8h x0 = *reinterpret_cast<const v8h*>(kp);
            v8h x1 = *reinterpret_cast<const v8h*>(kp + 16);
            kb0 = __builtin_shufflevector(x0, x1, 0, 1, 2, 3, 4, 5, 6, 7,
                                          8, 9, 10, 11, 12, 13, 14, 15);
            const f16* kp2 = Km + (size_t)(b * Sk + k0 + 16 + r) * DMODEL
                             + h * 32 + hi * 8;
            v8h y0 = *reinterpret_cast<const v8h*>(kp2);
            v8h y1 = *reinterpret_cast<const v8h*>(kp2 + 16);
            kb1 = __builtin_shufflevector(y0, y1, 0, 1, 2, 3, 4, 5, 6, 7,
                                          8, 9, 10, 11, 12, 13, 14, 15);
        }
        // ---- scores: 4 WMMAs (2 q-tiles x 2 key-tiles) --------------------
        v8f s[2][2];
        #pragma unroll
        for (int qt = 0; qt < 2; ++qt) {
            s[qt][0] = __builtin_amdgcn_wmma_f32_16x16x32_f16(
                false, aq[qt], false, kb0, (short)0, (v8f){}, false, false);
            s[qt][1] = __builtin_amdgcn_wmma_f32_16x16x32_f16(
                false, aq[qt], false, kb1, (short)0, (v8f){}, false, false);
        }

        // ---- online softmax over the 32 new columns, per q-tile ----------
        const float mk0 = mrow[k0 + r];
        const float mk1 = mrow[k0 + 16 + r];
        #pragma unroll
        for (int qt = 0; qt < 2; ++qt) {
            float p0[8], p1[8], mx[8];
            #pragma unroll
            for (int i = 0; i < 8; ++i) {
                p0[i] = s[qt][0][i] * scale + mk0;
                p1[i] = s[qt][1][i] * scale + mk1;
                mx[i] = fmaxf(p0[i], p1[i]);
            }
            #pragma unroll
            for (int i = 0; i < 8; ++i) mx[i] = redmax16(mx[i]);

            float rs[8];
            #pragma unroll
            for (int i = 0; i < 8; ++i) {
                const float mn = fmaxf(m_i[qt][i], mx[i]);
                const float corr = __expf(m_i[qt][i] - mn);
                m_i[qt][i] = mn;
                p0[i] = __expf(p0[i] - mn);
                p1[i] = __expf(p1[i] - mn);
                rs[i] = p0[i] + p1[i];
                l_i[qt][i] = l_i[qt][i] * corr;   // + rs after reduction
                o[qt][0][i] *= corr;
                o[qt][1][i] *= corr;
            }
            #pragma unroll
            for (int i = 0; i < 8; ++i) {
                rs[i] = redsum16(rs[i]);
                l_i[qt][i] += rs[i];
            }
            // probs: C layout -> LDS (re-laid out as A fragment below)
            #pragma unroll
            for (int i = 0; i < 8; ++i) {
                P[qt * 512 + (hi * 8 + i) * 32 + r]      = (f16)p0[i];
                P[qt * 512 + (hi * 8 + i) * 32 + 16 + r] = (f16)p1[i];
            }
        }
        __syncthreads();

        // ---- V fragments (shared by both q-tiles) -------------------------
        v16h vb0, vb1;
        #pragma unroll
        for (int j = 0; j < 16; ++j) {
            const f16* vp = Vm + (size_t)(b * Sk + k0 + hi * 8 + j + (j >= 8 ? 8 : 0)) * DMODEL
                            + h * 32;
            vb0[j] = vp[r];
            vb1[j] = vp[16 + r];
        }
        // ---- ctx += P @ V : 4 WMMAs --------------------------------------
        #pragma unroll
        for (int qt = 0; qt < 2; ++qt) {
            const f16* pp = &P[qt * 512 + r * 32 + hi * 8];
            v8h plo = *reinterpret_cast<const v8h*>(pp);
            v8h phi = *reinterpret_cast<const v8h*>(pp + 16);
            v16h pa = __builtin_shufflevector(plo, phi, 0, 1, 2, 3, 4, 5, 6, 7,
                                              8, 9, 10, 11, 12, 13, 14, 15);
            o[qt][0] = __builtin_amdgcn_wmma_f32_16x16x32_f16(
                false, pa, false, vb0, (short)0, o[qt][0], false, false);
            o[qt][1] = __builtin_amdgcn_wmma_f32_16x16x32_f16(
                false, pa, false, vb1, (short)0, o[qt][1], false, false);
        }
        __syncthreads();
    }

    // ---- normalize and store ctx (b, q, h*32+d) ---------------------------
    #pragma unroll
    for (int qt = 0; qt < 2; ++qt) {
        #pragma unroll
        for (int i = 0; i < 8; ++i) {
            const float inv = 1.0f / l_i[qt][i];
            const size_t row =
                (size_t)(b * Sq + q0 + qt * 16 + hi * 8 + i) * DMODEL + h * 32;
            ctx[row + r]      = (f16)(o[qt][0][i] * inv);
            ctx[row + 16 + r] = (f16)(o[qt][1][i] * inv);
        }
    }
}

// ---------------------------------------------------------------------------
// out = LayerNorm(a + b) * g + beta ; also emits f16 copy for next GEMM.
// One 256-thread block per row of 256 elements.
// ---------------------------------------------------------------------------
__global__ __launch_bounds__(256)
void add_ln_kernel(const float* __restrict__ a, const float* __restrict__ bsrc,
                   const float* __restrict__ g, const float* __restrict__ beta,
                   float* __restrict__ out32, f16* __restrict__ out16)
{
    __shared__ float red[8];
    const int row = blockIdx.x;
    const int t   = threadIdx.x;
    const size_t idx = (size_t)row * DMODEL + t;

    const float x = a[idx] + bsrc[idx];

    float s = x;
    #pragma unroll
    for (int d = 1; d < 32; d <<= 1) s += __shfl_xor(s, d, 32);
    if ((t & 31) == 0) red[t >> 5] = s;
    __syncthreads();
    float tot = 0.0f;
    #pragma unroll
    for (int w = 0; w < 8; ++w) tot += red[w];
    const float mean = tot * (1.0f / DMODEL);
    __syncthreads();

    const float dv = x - mean;
    float s2 = dv * dv;
    #pragma unroll
    for (int d = 1; d < 32; d <<= 1) s2 += __shfl_xor(s2, d, 32);
    if ((t & 31) == 0) red[t >> 5] = s2;
    __syncthreads();
    float tot2 = 0.0f;
    #pragma unroll
    for (int w = 0; w < 8; ++w) tot2 += red[w];
    const float var = tot2 * (1.0f / DMODEL);

    const float y = dv * rsqrtf(var + LN_EPS) * g[t] + beta[t];
    out32[idx] = y;
    if (out16) out16[idx] = (f16)y;
}

// ---------------------------------------------------------------------------
// Orchestration
// ---------------------------------------------------------------------------
extern "C" void kernel_launch(void* const* d_in, const int* in_sizes, int n_in,
                              void* d_out, int out_size, void* d_ws, size_t ws_size,
                              hipStream_t stream)
{
    (void)in_sizes; (void)n_in; (void)out_size; (void)ws_size;

    const int B = 4, Sq = 2048, Sk = 1024, FFN = 2048;
    const int Mq = B * Sq;     // 8192 query rows
    const int Mt = B * Sk;     // 4096 text rows

    const float* vis      = (const float*)d_in[0];
    const float* txt      = (const float*)d_in[1];
    const float* vis_mask = (const float*)d_in[2];
    const float* txt_mask = (const float*)d_in[3];
    const float* sa_wq = (const float*)d_in[4];  const float* sa_bq = (const float*)d_in[5];
    const float* sa_wk = (const float*)d_in[6];  const float* sa_bk = (const float*)d_in[7];
    const float* sa_wv = (const float*)d_in[8];  const float* sa_bv = (const float*)d_in[9];
    const float* sa_wo = (const float*)d_in[10]; const float* sa_bo = (const float*)d_in[11];
    const float* ca_wq = (const float*)d_in[12]; const float* ca_bq = (const float*)d_in[13];
    const float* ca_wk = (const float*)d_in[14]; const float* ca_bk = (const float*)d_in[15];
    const float* ca_wv = (const float*)d_in[16]; const float* ca_bv = (const float*)d_in[17];
    const float* ca_wo = (const float*)d_in[18]; const float* ca_bo = (const float*)d_in[19];
    const float* ln1_g = (const float*)d_in[20]; const float* ln1_b = (const float*)d_in[21];
    const float* ln2_g = (const float*)d_in[22]; const float* ln2_b = (const float*)d_in[23];
    const float* ln3_g = (const float*)d_in[24]; const float* ln3_b = (const float*)d_in[25];
    const float* ln4_g = (const float*)d_in[26]; const float* ln4_b = (const float*)d_in[27];
    const float* fc1_w = (const float*)d_in[28]; const float* fc1_b = (const float*)d_in[29];
    const float* fc2_w = (const float*)d_in[30]; const float* fc2_b = (const float*)d_in[31];

    char* ws = (char*)d_ws;
    const size_t MB = 1024ull * 1024ull;
    f16*   q_h    = (f16*)  (ws + 0 * MB);    // 4 MB  (8192 x 256)
    f16*   k_h    = (f16*)  (ws + 4 * MB);    // 4 MB
    f16*   v_h    = (f16*)  (ws + 8 * MB);    // 4 MB
    f16*   ctx_h  = (f16*)  (ws + 12 * MB);   // 4 MB
    f16*   xin_h  = (f16*)  (ws + 16 * MB);   // 4 MB  (current activation, f16)
    f16*   txt_h  = (f16*)  (ws + 20 * MB);   // 2 MB
    float* tmp_f  = (float*)(ws + 24 * MB);   // 8 MB  (GEMM f32 output)
    float* hA     = (float*)(ws + 32 * MB);   // 8 MB
    float* hB     = (float*)(ws + 40 * MB);   // 8 MB
    f16*   mid_h  = (f16*)  (ws + 48 * MB);   // 32 MB (8192 x 2048)

    const int nvis = Mq * DMODEL, ntxt = Mt * DMODEL;
    f32_to_f16_kernel<<<(nvis + 255) / 256, 256, 0, stream>>>(vis, xin_h, nvis);
    f32_to_f16_kernel<<<(ntxt + 255) / 256, 256, 0, stream>>>(txt, txt_h, ntxt);

    const dim3 gP(DMODEL / 64, Mq / 64);     // (4, 128) projections on queries
    const dim3 gT(DMODEL / 64, Mt / 64);     // (4, 64)  projections on text
    const dim3 gF1(FFN / 64, Mq / 64);       // (32, 128) fc1
    const dim3 gAttn(Sq / 32, 8, B);         // (64, 8, 4): 32 q-rows per wave

    // ---- self attention ---------------------------------------------------
    gemm_kernel<true, false><<<gP, 128, 0, stream>>>(xin_h, sa_wq, sa_bq, q_h, Mq, DMODEL, DMODEL);
    gemm_kernel<true, false><<<gP, 128, 0, stream>>>(xin_h, sa_wk, sa_bk, k_h, Mq, DMODEL, DMODEL);
    gemm_kernel<true, false><<<gP, 128, 0, stream>>>(xin_h, sa_wv, sa_bv, v_h, Mq, DMODEL, DMODEL);
    attn_kernel<<<gAttn, 32, 0, stream>>>(q_h, k_h, v_h, vis_mask, ctx_h, Sq, Sq);
    gemm_kernel<false, false><<<gP, 128, 0, stream>>>(ctx_h, sa_wo, sa_bo, tmp_f, Mq, DMODEL, DMODEL);
    add_ln_kernel<<<Mq, 256, 0, stream>>>(vis, tmp_f, ln1_g, ln1_b, hA, xin_h);

    // ---- text cross attention --------------------------------------------
    gemm_kernel<true, false><<<gP, 128, 0, stream>>>(xin_h, ca_wq, ca_bq, q_h, Mq, DMODEL, DMODEL);
    gemm_kernel<true, false><<<gT, 128, 0, stream>>>(txt_h, ca_wk, ca_bk, k_h, Mt, DMODEL, DMODEL);
    gemm_kernel<true, false><<<gT, 128, 0, stream>>>(txt_h, ca_wv, ca_bv, v_h, Mt, DMODEL, DMODEL);
    attn_kernel<<<gAttn, 32, 0, stream>>>(q_h, k_h, v_h, txt_mask, ctx_h, Sq, Sk);
    gemm_kernel<false, false><<<gP, 128, 0, stream>>>(ctx_h, ca_wo, ca_bo, tmp_f, Mq, DMODEL, DMODEL);
    add_ln_kernel<<<Mq, 256, 0, stream>>>(hA, tmp_f, ln2_g, ln2_b, hB, xin_h);

    // ---- third block: LN(h + h) ------------------------------------------
    add_ln_kernel<<<Mq, 256, 0, stream>>>(hB, hB, ln3_g, ln3_b, hA, xin_h);

    // ---- FFN --------------------------------------------------------------
    gemm_kernel<true, true><<<gF1, 128, 0, stream>>>(xin_h, fc1_w, fc1_b, mid_h, Mq, FFN, DMODEL);
    gemm_kernel<false, false><<<gP, 128, 0, stream>>>(mid_h, fc2_w, fc2_b, tmp_f, Mq, DMODEL, FFN);
    add_ln_kernel<<<Mq, 256, 0, stream>>>(hA, tmp_f, ln4_g, ln4_b, (float*)d_out, (f16*)nullptr);
}
